// SpatialNTM_11536282157641
// MI455X (gfx1250) — matrix operand order repaired and checked by reading, not compiled, not run on hardware
//
#include <hip/hip_runtime.h>
#include <hip/hip_bf16.h>
#include <math.h>

// ---------------------------------------------------------------------------
// SpatialNTM fused pipeline for MI455X (gfx1250, wave32).
//
// Roofline: dominated by ~670MB HBM traffic over c_prev/c_xf/c_new
// (~29us @ 23.3TB/s). The dense projections (Wk/We/Wv/W_ih/W_hh) are tiny
// (~0.15 GFLOP) and run on v_wmma_f32_16x16x32_f16 with f32 accumulation.
// Operands with ragged K/N are pre-padded so the WMMA inner loop is
// branchless: 4x global_load_b128 per fragment, pack with v_cvt_pk_rtz.
// ---------------------------------------------------------------------------

typedef __attribute__((ext_vector_type(16))) _Float16 v16h;
typedef __attribute__((ext_vector_type(2)))  _Float16 v2h;
typedef __attribute__((ext_vector_type(8)))  float    v8f;

#define NB    64
#define C1    4096
#define C2    128
#define DIMH  512
#define HID   516
#define INP   132
#define G3    1548
#define G3P   1552      // N padded to 16
#define KIP   160       // INP padded to 32
#define KHP   544       // HID padded to 32
#define EPSF  1e-8f

__device__ __forceinline__ float sigmoidf_(float x) { return 1.0f / (1.0f + expf(-x)); }

// v_cvt_pk_rtz_f16_f32: pack two f32 into two f16 (builtin returns __fp16x2;
// bit-identical to _Float16x2, so bit_cast).
__device__ __forceinline__ v2h cvt2_(float x, float y) {
    return __builtin_bit_cast(v2h, __builtin_amdgcn_cvt_pkrtz(x, y));
}

__device__ __forceinline__ void pack8_(v16h& d, int base, float4 u, float4 w) {
    v2h p0 = cvt2_(u.x, u.y);
    v2h p1 = cvt2_(u.z, u.w);
    v2h p2 = cvt2_(w.x, w.y);
    v2h p3 = cvt2_(w.z, w.w);
    d[base + 0] = p0[0]; d[base + 1] = p0[1];
    d[base + 2] = p1[0]; d[base + 3] = p1[1];
    d[base + 4] = p2[0]; d[base + 5] = p2[1];
    d[base + 6] = p3[0]; d[base + 7] = p3[1];
}

// ---------------------------------------------------------------------------
// WMMA GEMM:  Out[m,j] = act( sum_k A[m,k] * Wt[j,k] + bias[j] )
// One wave per 16x16 tile. REQUIREMENTS (guaranteed by caller via padding):
//   K multiple of 32; A has >= K readable cols per row (lda*4 % 16 == 0);
//   Wt has >= gridDim.y*16 rows and >= K cols (ldw*4 % 16 == 0).
// Fragment layouts per CDNA5 ISA 7.12.2 (wave32):
//   A: lane = M%16 (both halves), elem e -> K = (e>>3)*16 + half*8 + (e&7)
//      => two contiguous float8 runs: [kk+half*8, +8) and [kk+16+half*8, +8)
//   B: lane = N%16 (both halves), elem e -> K = (e>>3)*16 + half*8 + (e&7)
//      => two contiguous float8 runs: [kk+half*8, +8) and [kk+16+half*8, +8)
//   C/D: lane -> col = lane%16, VGPR i -> row = i + 8*half
// ---------------------------------------------------------------------------
__global__ void gemm_wmma_f16(const float* __restrict__ A, int lda,
                              const float* __restrict__ Wt, int ldw, int Nout, int K,
                              const float* __restrict__ bias,
                              float* __restrict__ Out, int ldo, int act)
{
    const int lane = threadIdx.x & 31;
    const int half = lane >> 4;
    const int l15  = lane & 15;
    const int mt   = blockIdx.x;
    const int nt   = blockIdx.y;

    const float* Arow = A  + (size_t)(mt * 16 + l15) * lda;
    const float* Brow = Wt + (size_t)(nt * 16 + l15) * ldw;

    v8f acc = {};
    for (int kk = 0; kk < K; kk += 32) {
        const float4* pa0 = reinterpret_cast<const float4*>(Arow + kk + half * 8);
        const float4* pa1 = reinterpret_cast<const float4*>(Arow + kk + 16 + half * 8);
        const float4* pb  = reinterpret_cast<const float4*>(Brow + kk + half * 8);
        const float4* pb1 = reinterpret_cast<const float4*>(Brow + kk + 16 + half * 8);
        float4 a0 = pa0[0], a1 = pa0[1];
        float4 a2 = pa1[0], a3 = pa1[1];
        float4 b0 = pb[0],  b1 = pb[1], b2 = pb1[0], b3 = pb1[1];

        v16h a, b;
        pack8_(a, 0, a0, a1);
        pack8_(a, 8, a2, a3);
        pack8_(b, 0, b0, b1);
        pack8_(b, 8, b2, b3);

        acc = __builtin_amdgcn_wmma_f32_16x16x32_f16(
            /*neg_a=*/false, a, /*neg_b=*/false, b,
            /*c_mod=*/(short)0, acc, /*reuse_a=*/false, /*reuse_b=*/false);
    }

    const int col = nt * 16 + l15;
    if (col < Nout) {
        const float bsv = bias[col];
#pragma unroll
        for (int i = 0; i < 8; ++i) {
            const int row = mt * 16 + i + (half << 3);
            float o = acc[i] + bsv;
            if (act == 1) o = sigmoidf_(o);
            Out[(size_t)row * ldo + col] = o;
        }
    }
}

// Zero-padded 2D copy: dst[drows x dcols] <- src[srows x scols], pad = 0.
__global__ void pad_copy(const float* __restrict__ src, int srows, int scols,
                         float* __restrict__ dst, int drows, int dcols)
{
    const int idx = blockIdx.x * blockDim.x + threadIdx.x;
    if (idx >= drows * dcols) return;
    const int r0 = idx / dcols, c0 = idx % dcols;
    dst[idx] = (r0 < srows && c0 < scols) ? src[(size_t)r0 * scols + c0] : 0.0f;
}

// ---------------------------------------------------------------------------
// beta[n] = softplus-style transform of origin_h[n] . Wb + bb
// ---------------------------------------------------------------------------
__global__ void beta_kernel(const float* __restrict__ h_prev,
                            const float* __restrict__ Wb,
                            const float* __restrict__ bb,
                            float* __restrict__ beta)
{
    const int n = blockIdx.x, lane = threadIdx.x;
    float s = 0.f;
    for (int k = lane; k < DIMH; k += 32) s += h_prev[(size_t)n * HID + k] * Wb[k];
    for (int off = 16; off; off >>= 1) s += __shfl_xor(s, off, 32);
    if (lane == 0) {
        float x  = s + bb[0];
        float bp = fmaxf(x, 0.f), bn = fminf(x, 0.f);
        beta[n] = log1pf(expf(bn)) + bp + log1pf(expf(-bp)) + (1.0f - 0.69314718055994531f);
    }
}

// ---------------------------------------------------------------------------
// Cosine similarity of every c_prev / c_xf row against k[n]. One pass over
// both 134MB tensors. Block = 8 waves; wave per row (128 floats = lane*f4).
// ---------------------------------------------------------------------------
__global__ void sim_kernel(const float* __restrict__ c_prev,
                           const float* __restrict__ c_xf,
                           const float* __restrict__ k,
                           float* __restrict__ s_kc, float* __restrict__ s_kx)
{
    __shared__ __align__(16) float sk[C2];
    __shared__ float sknorm;
    const int n = blockIdx.y, t = threadIdx.x;
    const int lane = t & 31, wave = t >> 5;

    if (t < C2) sk[t] = k[(size_t)n * C2 + t];
    __syncthreads();
    if (t == 0) {
        float s = 0.f;
        for (int i = 0; i < C2; ++i) s += sk[i] * sk[i];
        sknorm = fmaxf(sqrtf(s), EPSF);
    }
    __syncthreads();

    const float4 kv = reinterpret_cast<const float4*>(sk)[lane];
    const int base = blockIdx.x * 64;
    for (int it = 0; it < 8; ++it) {
        const int c = base + it * 8 + wave;
        const size_t row = (size_t)n * C1 + c;
        const float4* pc = reinterpret_cast<const float4*>(c_prev + row * C2);
        const float4* px = reinterpret_cast<const float4*>(c_xf   + row * C2);
        if (it + 1 < 8) {  // gfx1250 global_prefetch_b8 for next row
            __builtin_prefetch(pc + 256, 0, 0);
            __builtin_prefetch(px + 256, 0, 0);
        }
        float4 a = pc[lane], x = px[lane];
        float dc = a.x * kv.x + a.y * kv.y + a.z * kv.z + a.w * kv.w;
        float nc = a.x * a.x + a.y * a.y + a.z * a.z + a.w * a.w;
        float dx = x.x * kv.x + x.y * kv.y + x.z * kv.z + x.w * kv.w;
        float nx = x.x * x.x + x.y * x.y + x.z * x.z + x.w * x.w;
        for (int off = 16; off; off >>= 1) {
            dc += __shfl_xor(dc, off, 32);
            nc += __shfl_xor(nc, off, 32);
            dx += __shfl_xor(dx, off, 32);
            nx += __shfl_xor(nx, off, 32);
        }
        if (lane == 0) {
            s_kc[row] = dc / (fmaxf(sqrtf(nc), EPSF) * sknorm);
            s_kx[row] = dx / (fmaxf(sqrtf(nx), EPSF) * sknorm);
        }
    }
}

// ---------------------------------------------------------------------------
// In-place softmax over 4096 elems of s*beta. grid = (N, 2): y picks array.
// ---------------------------------------------------------------------------
__global__ void softmax4096(float* __restrict__ s_kc, float* __restrict__ s_kx,
                            const float* __restrict__ beta)
{
    __shared__ float red[256];
    const int n = blockIdx.x, t = threadIdx.x;
    float* arr = ((blockIdx.y == 0) ? s_kc : s_kx) + (size_t)n * C1;
    const float b = beta[n];

    float xs[16], mx = -INFINITY;
#pragma unroll
    for (int i = 0; i < 16; ++i) {
        float x = arr[t + i * 256] * b;
        xs[i] = x;
        mx = fmaxf(mx, x);
    }
    red[t] = mx; __syncthreads();
    for (int off = 128; off; off >>= 1) {
        if (t < off) red[t] = fmaxf(red[t], red[t + off]);
        __syncthreads();
    }
    mx = red[0]; __syncthreads();

    float sum = 0.f;
#pragma unroll
    for (int i = 0; i < 16; ++i) { xs[i] = expf(xs[i] - mx); sum += xs[i]; }
    red[t] = sum; __syncthreads();
    for (int off = 128; off; off >>= 1) {
        if (t < off) red[t] += red[t + off];
        __syncthreads();
    }
    const float inv = 1.0f / red[0];
#pragma unroll
    for (int i = 0; i < 16; ++i) arr[t + i * 256] = xs[i] * inv;
}

// ---------------------------------------------------------------------------
// w_bias[n,c,b] = softmax_b of rolled w_kc neighborhood (SHIFTS).
// ---------------------------------------------------------------------------
__global__ void wbias_kernel(const float* __restrict__ w_kc, float* __restrict__ w_bias)
{
    const int idx = blockIdx.x * blockDim.x + threadIdx.x;   // < N*C1
    const int n = idx >> 12, c = idx & (C1 - 1);
    const int x = c >> 6, y = c & 63;
    const float* w = w_kc + (size_t)n * C1;
    float v0 = w[x * 64 + ((y + 63) & 63)];          // roll (0, 1)
    float v1 = w[((x + 63) & 63) * 64 + y];          // roll (1, 0)
    float v2 = w[x * 64 + ((y + 1) & 63)];           // roll (0,-1)
    float v3 = w[((x + 1) & 63) * 64 + y];           // roll (-1,0)
    float m  = fmaxf(fmaxf(v0, v1), fmaxf(v2, v3));
    float e0 = expf(v0 - m), e1 = expf(v1 - m), e2 = expf(v2 - m), e3 = expf(v3 - m);
    float inv = 1.0f / (e0 + e1 + e2 + e3);
    float4 out = make_float4(e0 * inv, e1 * inv, e2 * inv, e3 * inv);
    reinterpret_cast<float4*>(w_bias)[idx] = out;
}

// ---------------------------------------------------------------------------
// r[n,d] = sum_c w_kx[n,c] * concat(c_xf, w_bias)[n,c,d]  -- partials over
// c-chunks of 512 (grid = (8, N), block = 160 threads, t<132 active).
// ---------------------------------------------------------------------------
__global__ void r_partial(const float* __restrict__ w_kx, const float* __restrict__ c_xf,
                          const float* __restrict__ w_bias, float* __restrict__ r_part)
{
    const int n = blockIdx.y, chunk = blockIdx.x, t = threadIdx.x;
    const int c0 = chunk * 512;
    float acc = 0.f;
    if (t < C2) {
        for (int c = c0; c < c0 + 512; ++c) {
            const size_t row = (size_t)n * C1 + c;
            if (c + 1 < c0 + 512) __builtin_prefetch(c_xf + (row + 1) * C2 + t, 0, 0);
            acc += w_kx[row] * c_xf[row * C2 + t];
        }
    } else if (t < INP) {
        const int bdim = t - C2;
        for (int c = c0; c < c0 + 512; ++c) {
            const size_t row = (size_t)n * C1 + c;
            acc += w_kx[row] * w_bias[row * 4 + bdim];
        }
    }
    if (t < INP) r_part[((size_t)n * 8 + chunk) * INP + t] = acc;
}

// Reduce partials into K-padded r (64 x KIP): d < 132 summed, pad = 0.
__global__ void r_reduce(const float* __restrict__ r_part, float* __restrict__ rpad)
{
    const int idx = blockIdx.x * blockDim.x + threadIdx.x;
    if (idx >= NB * KIP) return;
    const int n = idx / KIP, d = idx % KIP;
    float s = 0.f;
    if (d < INP)
        for (int ch = 0; ch < 8; ++ch) s += r_part[((size_t)n * 8 + ch) * INP + d];
    rpad[idx] = s;
}

// ---------------------------------------------------------------------------
// GRU gate combine: h_o = (1-z)*n + z*h_prev
// ---------------------------------------------------------------------------
__global__ void gru_gate(const float* __restrict__ gi, const float* __restrict__ gh,
                         const float* __restrict__ h_prev, float* __restrict__ h_o)
{
    const int idx = blockIdx.x * blockDim.x + threadIdx.x;
    if (idx >= NB * HID) return;
    const int n = idx / HID, j = idx % HID;
    const float* gin = gi + (size_t)n * G3;
    const float* ghn = gh + (size_t)n * G3;
    float rg = sigmoidf_(gin[j]       + ghn[j]);
    float zg = sigmoidf_(gin[HID + j] + ghn[HID + j]);
    float ng = tanhf(gin[2 * HID + j] + rg * ghn[2 * HID + j]);
    h_o[idx] = (1.0f - zg) * ng + zg * h_prev[idx];
}

// h_o_bias[n,:] = softmax(h_o[n, 512:516])
__global__ void hob_kernel(const float* __restrict__ h_o, float* __restrict__ hob)
{
    const int n = threadIdx.x;
    if (n >= NB) return;
    const float* p = h_o + (size_t)n * HID + DIMH;
    float v0 = p[0], v1 = p[1], v2 = p[2], v3 = p[3];
    float m  = fmaxf(fmaxf(v0, v1), fmaxf(v2, v3));
    float e0 = expf(v0 - m), e1 = expf(v1 - m), e2 = expf(v2 - m), e3 = expf(v3 - m);
    float inv = 1.0f / (e0 + e1 + e2 + e3);
    hob[n * 4 + 0] = e0 * inv; hob[n * 4 + 1] = e1 * inv;
    hob[n * 4 + 2] = e2 * inv; hob[n * 4 + 3] = e3 * inv;
}

// ---------------------------------------------------------------------------
// Final memory update: c_new = c_prev*(1 - w_kc*e) + w_write*v
// grid = (C1/128, N), block = 256 (8 waves, wave per row, float4 per lane).
// ---------------------------------------------------------------------------
__global__ void cnew_kernel(const float* __restrict__ c_prev, const float* __restrict__ w_kc,
                            const float* __restrict__ w_bias, const float* __restrict__ hob,
                            const float* __restrict__ e, const float* __restrict__ v,
                            float* __restrict__ c_new)
{
    __shared__ __align__(16) float se[C2];
    __shared__ __align__(16) float sv[C2];
    __shared__ float shb[4];
    const int n = blockIdx.y, t = threadIdx.x;
    if (t < C2) { se[t] = e[(size_t)n * C2 + t]; sv[t] = v[(size_t)n * C2 + t]; }
    if (t < 4) shb[t] = hob[n * 4 + t];
    __syncthreads();

    const int lane = t & 31, wave = t >> 5;
    const float4 ev = reinterpret_cast<const float4*>(se)[lane];
    const float4 vv = reinterpret_cast<const float4*>(sv)[lane];
    const int base = blockIdx.x * 128;
    for (int it = 0; it < 16; ++it) {
        const int c = base + it * 8 + wave;
        const size_t row = (size_t)n * C1 + c;
        const float  wkc = w_kc[row];
        const float4 wb  = reinterpret_cast<const float4*>(w_bias)[row];
        const float  ww  = wb.x * shb[0] + wb.y * shb[1] + wb.z * shb[2] + wb.w * shb[3];
        const float4* pc = reinterpret_cast<const float4*>(c_prev + row * C2);
        if (it + 1 < 16) __builtin_prefetch(pc + 256, 0, 0);
        float4 a = pc[lane], o;
        o.x = a.x * (1.0f - wkc * ev.x) + ww * vv.x;
        o.y = a.y * (1.0f - wkc * ev.y) + ww * vv.y;
        o.z = a.z * (1.0f - wkc * ev.z) + ww * vv.z;
        o.w = a.w * (1.0f - wkc * ev.w) + ww * vv.w;
        reinterpret_cast<float4*>(c_new + row * C2)[lane] = o;
    }
}

// ---------------------------------------------------------------------------
extern "C" void kernel_launch(void* const* d_in, const int* in_sizes, int n_in,
                              void* d_out, int out_size, void* d_ws, size_t ws_size,
                              hipStream_t stream)
{
    const float* h_prev = (const float*)d_in[0];   // (64,516)
    const float* c_prev = (const float*)d_in[1];   // (64,4096,128)
    const float* c_xf   = (const float*)d_in[2];   // (64,4096,128)
    const float* Wk     = (const float*)d_in[3];   // (128,512)
    const float* bk     = (const float*)d_in[4];
    const float* Wb     = (const float*)d_in[5];   // (1,512)
    const float* bb     = (const float*)d_in[6];
    const float* We     = (const float*)d_in[7];   // (128,512)
    const float* be     = (const float*)d_in[8];
    const float* Wv     = (const float*)d_in[9];   // (128,512)
    const float* bv     = (const float*)d_in[10];
    const float* W_ih   = (const float*)d_in[11];  // (1548,132)
    const float* W_hh   = (const float*)d_in[12];  // (1548,516)
    const float* b_ih   = (const float*)d_in[13];
    const float* b_hh   = (const float*)d_in[14];

    float* h_o   = (float*)d_out;                  // (64,516)
    float* c_new = h_o + (size_t)NB * HID;         // (64,4096,128)

    // workspace layout (floats)
    float* ws      = (float*)d_ws;
    float* k       = ws;                                  // 64*128
    float* beta    = k       + NB * C2;                   // 64
    float* s_kc    = beta    + NB;                        // 64*4096 (-> w_kc)
    float* s_kx    = s_kc    + (size_t)NB * C1;           // 64*4096 (-> w_kx)
    float* w_bias  = s_kx    + (size_t)NB * C1;           // 64*4096*4
    float* r_part  = w_bias  + (size_t)NB * C1 * 4;       // 64*8*132
    float* rpad    = r_part  + (size_t)NB * 8 * INP;      // 64*160 (K-padded)
    float* gi      = rpad    + (size_t)NB * KIP;          // 64*1548
    float* gh      = gi      + (size_t)NB * G3;           // 64*1548
    float* hob     = gh      + (size_t)NB * G3;           // 64*4
    float* e       = hob     + NB * 4;                    // 64*128
    float* v       = e       + NB * C2;                   // 64*128
    float* hpad    = v       + NB * C2;                   // 64*544
    float* Wih_pad = hpad    + (size_t)NB * KHP;          // 1552*160
    float* Whh_pad = Wih_pad + (size_t)G3P * KIP;         // 1552*544

    // 0. zero-padded copies so WMMA inner loops are branchless
    pad_copy<<<(NB * KHP + 255) / 256, 256, 0, stream>>>(h_prev, NB, HID, hpad, NB, KHP);
    pad_copy<<<(G3P * KIP + 255) / 256, 256, 0, stream>>>(W_ih, G3, INP, Wih_pad, G3P, KIP);
    pad_copy<<<(G3P * KHP + 255) / 256, 256, 0, stream>>>(W_hh, G3, HID, Whh_pad, G3P, KHP);

    // 1. k = origin_h @ Wk.T + bk            (WMMA)
    gemm_wmma_f16<<<dim3(NB / 16, C2 / 16), 32, 0, stream>>>(
        h_prev, HID, Wk, DIMH, C2, DIMH, bk, k, C2, 0);

    // 2. beta
    beta_kernel<<<NB, 32, 0, stream>>>(h_prev, Wb, bb, beta);

    // 3. cosine similarities (one pass over c_prev + c_xf)
    sim_kernel<<<dim3(C1 / 64, NB), 256, 0, stream>>>(c_prev, c_xf, k, s_kc, s_kx);

    // 4. softmax over c for both -> w_kc, w_kx (in place)
    softmax4096<<<dim3(NB, 2), 256, 0, stream>>>(s_kc, s_kx, beta);

    // 5. shift neighborhood softmax -> w_bias
    wbias_kernel<<<(NB * C1) / 256, 256, 0, stream>>>(s_kc, w_bias);

    // 6. r = w_kx . concat(c_xf, w_bias)  (partials, then K-padded reduce)
    r_partial<<<dim3(8, NB), 160, 0, stream>>>(s_kx, c_xf, w_bias, r_part);
    r_reduce<<<(NB * KIP + 255) / 256, 256, 0, stream>>>(r_part, rpad);

    // 7. GRU projections (WMMA, padded operands) + gates
    gemm_wmma_f16<<<dim3(NB / 16, G3P / 16), 32, 0, stream>>>(
        rpad, KIP, Wih_pad, KIP, G3, KIP, b_ih, gi, G3, 0);
    gemm_wmma_f16<<<dim3(NB / 16, G3P / 16), 32, 0, stream>>>(
        hpad, KHP, Whh_pad, KHP, G3, KHP, b_hh, gh, G3, 0);
    gru_gate<<<(NB * HID + 255) / 256, 256, 0, stream>>>(gi, gh, h_prev, h_o);

    // 8. h_o_bias softmax
    hob_kernel<<<1, NB, 0, stream>>>(h_o, hob);

    // 9. e = sigmoid(h_o[:, :512] @ We.T + be), v = ... @ Wv.T + bv  (WMMA)
    gemm_wmma_f16<<<dim3(NB / 16, C2 / 16), 32, 0, stream>>>(
        h_o, HID, We, DIMH, C2, DIMH, be, e, C2, 1);
    gemm_wmma_f16<<<dim3(NB / 16, C2 / 16), 32, 0, stream>>>(
        h_o, HID, Wv, DIMH, C2, DIMH, bv, v, C2, 0);

    // 10. memory update -> c_new
    cnew_kernel<<<dim3(C1 / 128, NB), 256, 0, stream>>>(
        c_prev, s_kc, w_bias, hob, e, v, c_new);
}
